// ContrastiveLoss_25941602467999
// MI455X (gfx1250) — matrix operand order repaired and checked by reading, not compiled, not run on hardware
//
#include <hip/hip_runtime.h>
#include <hip/hip_bf16.h>
#include <math.h>
#include <stdint.h>

typedef __attribute__((ext_vector_type(16))) _Float16 v16h;
typedef __attribute__((ext_vector_type(8)))  _Float16 v8h;
typedef __attribute__((ext_vector_type(8)))  float    v8f;
typedef __attribute__((ext_vector_type(4)))  float    v4f;
typedef __attribute__((ext_vector_type(4)))  int      v4i;

#define D 256
#define INV_BETA (1.0f / 0.08f)
#define EPS_NORM 1e-8f
#define CSPLIT 8

// gfx1250 async global->LDS copy path (ASYNCcnt-tracked), guarded so the file
// still compiles on toolchains without the builtins.
#if defined(__gfx1250__) && __has_builtin(__builtin_amdgcn_global_load_async_to_lds_b128) && __has_builtin(__builtin_amdgcn_s_wait_asynccnt)
#define USE_ASYNC_LDS 1
#else
#define USE_ASYNC_LDS 0
#endif

// ---------------------------------------------------------------------------
// Kernel 1: row-normalize z = [x1; x2]; emit fp16 zn and per-row 1/norm.
// One wave (32 lanes) per row; each lane owns 8 contiguous floats.
// ---------------------------------------------------------------------------
__global__ __launch_bounds__(256) void k_normalize(
    const float* __restrict__ x1, const float* __restrict__ x2,
    _Float16* __restrict__ znh, float* __restrict__ rnorm, int n) {
  const int wavesPerBlock = blockDim.x >> 5;
  const int row  = blockIdx.x * wavesPerBlock + (threadIdx.x >> 5);
  const int lane = threadIdx.x & 31;
  const int twoN = 2 * n;
  if (row >= twoN) return;

  const float* src = (row < n) ? (x1 + (size_t)row * D)
                               : (x2 + (size_t)(row - n) * D);
  v4f a = *(const v4f*)(src + lane * 8);
  v4f b = *(const v4f*)(src + lane * 8 + 4);
  float ss = 0.f;
#pragma unroll
  for (int t = 0; t < 4; ++t) ss += a[t] * a[t] + b[t] * b[t];
#pragma unroll
  for (int m = 16; m >= 1; m >>= 1) ss += __shfl_xor(ss, m, 32);

  const float r = 1.0f / fmaxf(sqrtf(ss), EPS_NORM);
  v8h o;
#pragma unroll
  for (int t = 0; t < 4; ++t) {
    o[t]     = (_Float16)(a[t] * r);
    o[t + 4] = (_Float16)(b[t] * r);
  }
  *(v8h*)(znh + (size_t)row * D + lane * 8) = o;
  if (lane == 0) rnorm[row] = r;
}

// ---------------------------------------------------------------------------
// Kernel 2: positive-pair similarities pos_half[i] = <zn_i, zn_{i+n}> / beta.
// ---------------------------------------------------------------------------
__global__ __launch_bounds__(256) void k_pos(
    const float* __restrict__ x1, const float* __restrict__ x2,
    const float* __restrict__ rnorm, float* __restrict__ pos_half, int n) {
  const int wavesPerBlock = blockDim.x >> 5;
  const int row  = blockIdx.x * wavesPerBlock + (threadIdx.x >> 5);
  const int lane = threadIdx.x & 31;
  if (row >= n) return;

  const float* a = x1 + (size_t)row * D;
  const float* b = x2 + (size_t)row * D;
  v4f a0 = *(const v4f*)(a + lane * 8), a1 = *(const v4f*)(a + lane * 8 + 4);
  v4f b0 = *(const v4f*)(b + lane * 8), b1 = *(const v4f*)(b + lane * 8 + 4);
  float dot = 0.f;
#pragma unroll
  for (int t = 0; t < 4; ++t) dot += a0[t] * b0[t] + a1[t] * b1[t];
#pragma unroll
  for (int m = 16; m >= 1; m >>= 1) dot += __shfl_xor(dot, m, 32);

  if (lane == 0)
    pos_half[row] = dot * rnorm[row] * rnorm[row + n] * INV_BETA;
}

// ---------------------------------------------------------------------------
// Kernel 3: zero the denominator accumulators (ws is poisoned by harness).
// ---------------------------------------------------------------------------
__global__ void k_zero(float* __restrict__ p, int count) {
  int i = blockIdx.x * blockDim.x + threadIdx.x;
  if (i < count) p[i] = 0.f;
}

// ---------------------------------------------------------------------------
// LDS staging of one B tile (16 columns x K=256 of fp16 zn -> 8 KB), stored
// fragment-major so each lane's B fragment for k-step `ks` is the contiguous
// 32 bytes at  buf + (ks*32 + lane)*16.
// Cooperative: 128 threads x 4 x 16B chunks = 8 KB.
// ---------------------------------------------------------------------------
__device__ __forceinline__ void stage_tile(const _Float16* __restrict__ znh,
                                           _Float16* __restrict__ buf,
                                           int col0, int tid) {
#pragma unroll
  for (int j = 0; j < 4; ++j) {
    const int c    = tid * 4 + j;   // 0..511 16-byte chunks
    const int ks   = c >> 6;        // 8 k-steps
    const int rem  = c & 63;
    const int ln   = rem >> 1;      // destination "lane slot" 0..31
    const int part = rem & 1;       // low/high 16B of the 32B fragment
    const _Float16* gsrc = znh + (size_t)(col0 + (ln & 15)) * D
                           + (ln >> 4) * 8 + ks * 32 + part * 16;
    _Float16* ldst = buf + (ks * 32 + ln) * 16 + part * 8;
#if USE_ASYNC_LDS
    // Toolchain signature: generic v4i* src / v4i* dst + imm offset + imm cpol.
    __builtin_amdgcn_global_load_async_to_lds_b128(
        (v4i*)gsrc, (v4i*)ldst, /*imm offset*/ 0, /*cpol*/ 0);
#else
    *(v8h*)ldst = *(const v8h*)gsrc;
#endif
  }
}

__device__ __forceinline__ void stage_commit() {
#if USE_ASYNC_LDS
  __builtin_amdgcn_s_wait_asynccnt(0);
#endif
  __syncthreads();
}

// ---------------------------------------------------------------------------
// Kernel 4: fused  denom[i] += sum_j!=i exp(<zn_i,zn_j>/beta).
//
// Block = 128 threads = 4 waves = 128 rows; grid.y splits columns (CSPLIT).
// Each wave holds TWO 16-row A strips fully in registers (2 x 8 x v16h), so
// every B fragment (read once from the shared LDS stage) feeds two
// v_wmma_f32_16x16x32_f16. B tiles are double-buffered in LDS with async
// global->LDS copies overlapping the WMMA+exp work on the current tile.
//
// Fragment layouts per CDNA5 ISA 7.12.2 (wave32):
//   A 16x32 f16 : lane (l<16) row M=l holds K {0..7,16..23}; lane l+16 holds
//                 K {8..15,24..31}  -> two contiguous v8h loads per lane.
//   B 32x16 f16 : lane (l<16) col N=l holds K {0..7,16..23}; lane l+16 holds K {8..15,24..31}
//                 (B = zn^T -> contiguous zn row segments, pre-staged in LDS).
//   C 16x16 f32 : VGPR r, lanes 0..15 -> (M=r, N=lane); lanes 16..31 -> M=r+8.
// ---------------------------------------------------------------------------
__global__ __launch_bounds__(128) void k_sim_rowsum(
    const _Float16* __restrict__ znh, float* __restrict__ denom, int n) {
  __shared__ __align__(32) _Float16 bstage[2][16 * D];   // 2 x 8 KB

  const int tid  = threadIdx.x;
  const int lane = tid & 31;
  const int wave = tid >> 5;
  const int twoN = 2 * n;

  const int rowBase      = blockIdx.x * 128 + wave * 32;  // 2 tiles per wave
  const int colsPerChunk = twoN / CSPLIT;                 // gridDim.y == CSPLIT
  const int colBase      = blockIdx.y * colsPerChunk;
  const int nColTiles    = colsPerChunk >> 4;

  const int half = lane >> 4;   // 0: first K/M half, 1: second
  const int mrow = lane & 15;

  // Preload both A strips (K = 0..255 in steps of 32); register-resident.
  v16h afrag[2][8];
#pragma unroll
  for (int rt = 0; rt < 2; ++rt) {
    const _Float16* abase =
        znh + (size_t)(rowBase + rt * 16 + mrow) * D + half * 8;
#pragma unroll
    for (int ks = 0; ks < 8; ++ks) {
      v8h lo = *(const v8h*)(abase + ks * 32);
      v8h hi = *(const v8h*)(abase + ks * 32 + 16);
#pragma unroll
      for (int t = 0; t < 8; ++t) {
        afrag[rt][ks][t]     = lo[t];
        afrag[rt][ks][t + 8] = hi[t];
      }
    }
  }

  float psum[2][8];
#pragma unroll
  for (int rt = 0; rt < 2; ++rt)
#pragma unroll
    for (int r = 0; r < 8; ++r) psum[rt][r] = 0.f;

  // Prologue: stage first B tile.
  stage_tile(znh, bstage[0], colBase, tid);
  stage_commit();

  for (int ct = 0; ct < nColTiles; ++ct) {
    const _Float16* cur = bstage[ct & 1];

    // Kick off the async copy of the next tile into the other buffer; it
    // overlaps the ds_load + WMMA + exp work below.
    if (ct + 1 < nColTiles)
      stage_tile(znh, bstage[(ct + 1) & 1], colBase + (ct + 1) * 16, tid);

    v8f c0 = {}, c1 = {};
#pragma unroll
    for (int ks = 0; ks < 8; ++ks) {
      v16h bfrag = *(const v16h*)(cur + (ks * 32 + lane) * 16);
      c0 = __builtin_amdgcn_wmma_f32_16x16x32_f16(
          false, afrag[0][ks], false, bfrag, (short)0, c0, false, false);
      c1 = __builtin_amdgcn_wmma_f32_16x16x32_f16(
          false, afrag[1][ks], false, bfrag, (short)0, c1, false, false);
    }

    // Epilogue: exp(sim/beta) with diagonal masked, fold into row sums.
    const int gn = colBase + ct * 16 + mrow;
#pragma unroll
    for (int r = 0; r < 8; ++r) {
      const int gm0 = rowBase + half * 8 + r;
      const int gm1 = gm0 + 16;
      psum[0][r] += (gm0 == gn) ? 0.f : __expf(c0[r] * INV_BETA);
      psum[1][r] += (gm1 == gn) ? 0.f : __expf(c1[r] * INV_BETA);
    }

    stage_commit();   // async done + all waves finished reading `cur`
  }

  // Reduce each row across its 16-lane group (xor masks 1,2,4,8 stay within
  // the half-wave), then one atomicAdd per row, distributed across lanes.
#pragma unroll
  for (int rt = 0; rt < 2; ++rt) {
#pragma unroll
    for (int r = 0; r < 8; ++r) {
      float s = psum[rt][r];
      s += __shfl_xor(s, 1, 32);
      s += __shfl_xor(s, 2, 32);
      s += __shfl_xor(s, 4, 32);
      s += __shfl_xor(s, 8, 32);
      if (mrow == r)
        atomicAdd(&denom[rowBase + rt * 16 + half * 8 + r], s);
    }
  }
}

// ---------------------------------------------------------------------------
// Kernel 5: loss = mean_i( log(denom_i) - pos_i ), pos repeats over halves.
// ---------------------------------------------------------------------------
__global__ __launch_bounds__(256) void k_loss(
    const float* __restrict__ denom, const float* __restrict__ pos_half,
    float* __restrict__ out, int n) {
  __shared__ float red[256];
  const int tid  = threadIdx.x;
  const int twoN = 2 * n;
  float acc = 0.f;
  for (int i = tid; i < twoN; i += blockDim.x) {
    const float p = pos_half[(i < n) ? i : (i - n)];
    acc += logf(denom[i]) - p;
  }
  red[tid] = acc;
  __syncthreads();
  for (int s = blockDim.x >> 1; s > 0; s >>= 1) {
    if (tid < s) red[tid] += red[tid + s];
    __syncthreads();
  }
  if (tid == 0) out[0] = red[0] / (float)twoN;
}

// ---------------------------------------------------------------------------
extern "C" void kernel_launch(void* const* d_in, const int* in_sizes, int n_in,
                              void* d_out, int out_size, void* d_ws, size_t ws_size,
                              hipStream_t stream) {
  (void)n_in; (void)out_size; (void)ws_size;
  const float* x1 = (const float*)d_in[0];
  const float* x2 = (const float*)d_in[1];
  float* out = (float*)d_out;

  const int n    = in_sizes[0] / D;   // 4096
  const int twoN = 2 * n;             // 8192

  // Workspace layout
  char* ws = (char*)d_ws;
  _Float16* znh = (_Float16*)ws;                       // 2N * D fp16 (4 MB)
  size_t znhBytes = (size_t)twoN * D * sizeof(_Float16);
  float* rnorm    = (float*)(ws + znhBytes);           // 2N f32
  float* pos_half = rnorm + twoN;                      // N  f32
  float* denom    = pos_half + n;                      // 2N f32

  // 1) normalize (8 waves / block -> one row per wave)
  k_normalize<<<twoN / 8, 256, 0, stream>>>(x1, x2, znh, rnorm, n);
  // 2) positive pairs
  k_pos<<<n / 8, 256, 0, stream>>>(x1, x2, rnorm, pos_half, n);
  // 3) clear denominators
  k_zero<<<(twoN + 255) / 256, 256, 0, stream>>>(denom, twoN);
  // 4) fused WMMA similarity + exp + masked row-sum
  {
    dim3 grid(twoN / 128, CSPLIT);   // 4 waves x 2 row-tiles each = 128 rows
    k_sim_rowsum<<<grid, 128, 0, stream>>>(znh, denom, n);
  }
  // 5) scalar loss
  k_loss<<<1, 256, 0, stream>>>(denom, pos_half, out, n);
}
